// Gpt2Block_22900765622611
// MI455X (gfx1250) — compile-verified, matrix operand order checked by reading, not run on hardware
//
#include <hip/hip_runtime.h>
#include <hip/hip_bf16.h>
#include <cstdint>

#define DEVFN __device__ __forceinline__

typedef __attribute__((ext_vector_type(16))) _Float16 v16h;
typedef __attribute__((ext_vector_type(8)))  _Float16 v8h;
typedef __attribute__((ext_vector_type(8)))  float    v8f;

// Problem constants (GPT-2 block)
constexpr int Bb  = 4;
constexpr int Ss  = 2048;
constexpr int Ee  = 2048;
constexpr int Hh  = 16;
constexpr int HDd = 128;
constexpr int Mm  = 8192;
constexpr int TOK = Bb * Ss;     // 8192 tokens
constexpr int E3  = 3 * Ee;      // 6144
constexpr float QSCALE = 0.08838834764831845f;   // 1/sqrt(128)
constexpr float GELU_C = 0.7978845608028654f;

// ---------------------------------------------------------------------------
// WMMA helpers
// ---------------------------------------------------------------------------
DEVFN v8f wmma_f16(v16h a, v16h b, v8f c) {
  return __builtin_amdgcn_wmma_f32_16x16x32_f16(false, a, false, b,
                                                (short)0, c, false, false);
}

// Load a 16x32 f16 A-layout fragment.  `base` points at (row0, k0) of the
// fragment; `ld` is the row stride in halves.  Per CDNA5 ISA:
//   lanes 0-15 : M = lane,    K = {k0..k0+7, k0+16..k0+23}
//   lanes 16-31: M = lane-16, K = {k0+8..k0+15, k0+24..k0+31}
DEVFN v16h load_frag(const _Float16* base, int ld) {
  const int lane = threadIdx.x & 31;
  const _Float16* p = base + (size_t)(lane & 15) * ld + ((lane & 16) ? 8 : 0);
  v8h lo = *(const v8h*)p;          // K window +0..7
  v8h hi = *(const v8h*)(p + 16);   // K window +16..23
  v16h r;
#pragma unroll
  for (int i = 0; i < 8; ++i) { r[i] = lo[i]; r[i + 8] = hi[i]; }
  return r;
}

// Generic shared pointers encode the LDS byte offset in their low 32 bits
// (generic = {SHARED_BASE aperture, 32-bit LDS offset} per ISA 10.2).
DEVFN uint32_t lds_addr_of(const void* p) {
  return (uint32_t)(uintptr_t)p;
}

// CDNA5 async DMA: copy 16B global -> LDS, tracked by ASYNCcnt (no VGPR
// round-trip, overlaps with WMMA compute).
DEVFN void async_copy_b128(uint32_t lds_addr, const _Float16* gptr) {
  asm volatile("global_load_async_to_lds_b128 %0, %1, off"
               :: "v"(lds_addr), "v"(gptr)
               : "memory");
}

// ---------------------------------------------------------------------------
// Weight conversion with transpose: in f32 [K][N] -> out f16 [N][K].
// Tiled through LDS so both sides stay coalesced.
// ---------------------------------------------------------------------------
__global__ __launch_bounds__(256) void cvt_transpose(
    const float* __restrict__ in, _Float16* __restrict__ out, int K, int N) {
  __shared__ float tile[32][33];
  const int n0 = blockIdx.x * 32;
  const int k0 = blockIdx.y * 32;
  const int tx = threadIdx.x & 31;
  const int ty = threadIdx.x >> 5;      // 8 rows per pass
#pragma unroll
  for (int i = 0; i < 32; i += 8)
    tile[ty + i][tx] = in[(size_t)(k0 + ty + i) * N + n0 + tx];
  __syncthreads();
#pragma unroll
  for (int i = 0; i < 32; i += 8)
    out[(size_t)(n0 + ty + i) * K + k0 + tx] = (_Float16)tile[tx][ty + i];
}

// ---------------------------------------------------------------------------
// LayerNorm: one block per row of E=2048, f32 in -> f16 out
// ---------------------------------------------------------------------------
__global__ __launch_bounds__(256) void ln_kernel(const float* __restrict__ x,
                                                 const float* __restrict__ g,
                                                 const float* __restrict__ b,
                                                 _Float16* __restrict__ out) {
  __shared__ float rs[256];
  __shared__ float rs2[256];
  const int row = blockIdx.x;
  const int tid = threadIdx.x;
  const float* xr = x + (size_t)row * Ee;
  float v[8];
  float s = 0.f, s2 = 0.f;
#pragma unroll
  for (int i = 0; i < 8; ++i) {
    v[i] = xr[tid + i * 256];
    s += v[i];
    s2 += v[i] * v[i];
  }
  rs[tid] = s; rs2[tid] = s2;
  __syncthreads();
  for (int off = 128; off > 0; off >>= 1) {
    if (tid < off) { rs[tid] += rs[tid + off]; rs2[tid] += rs2[tid + off]; }
    __syncthreads();
  }
  const float mu   = rs[0] * (1.0f / Ee);
  const float var  = rs2[0] * (1.0f / Ee) - mu * mu;
  const float rstd = rsqrtf(var + 1e-5f);
  _Float16* orow = out + (size_t)row * Ee;
#pragma unroll
  for (int i = 0; i < 8; ++i) {
    const int c = tid + i * 256;
    orow[c] = (_Float16)((v[i] - mu) * rstd * g[c] + b[c]);
  }
}

// ---------------------------------------------------------------------------
// WMMA GEMM:  C[M x N] = A[M x K](f16) * Bt[N x K](f16)^T + bias, epilogues.
// Block tile 128x128, 8 waves (4 row-waves x 2 col-waves), wave tile 32x64.
// A and Bt tiles staged into double-buffered LDS via async B128 DMA copies
// (ASYNCcnt), software-pipelined so tile i+1 streams while tile i computes.
// ---------------------------------------------------------------------------
constexpr int BM = 128, BN = 128, BK = 32, LDP = 40;
enum { EP_QKV = 0, EP_RES = 1, EP_GELU = 2 };

__global__ __launch_bounds__(256) void gemm_wmma_kernel(
    const _Float16* __restrict__ A, const _Float16* __restrict__ Bt,
    const float* __restrict__ bias, int Nr, int Kr, int mode,
    const float* __restrict__ res, float* __restrict__ outF,
    _Float16* __restrict__ outH, _Float16* __restrict__ vT) {
  __shared__ _Float16 As[2][BM * LDP];
  __shared__ _Float16 Bs[2][BN * LDP];

  const int tid  = threadIdx.x;
  const int lane = tid & 31;
  const int wave = tid >> 5;
  const int wr   = wave & 3;   // 4 row waves * 32 rows
  const int wc   = wave >> 2;  // 2 col waves * 64 cols
  const int m0   = blockIdx.y * BM;
  const int n0   = blockIdx.x * BN;

  // Per-thread staging geometry: 2 segments of 8 halves for each matrix.
  const int r0 = tid >> 1;                 // rows 0..127
  const int c0 = (tid & 1) * 16;           // 0 or 16
  const uint32_t ldsA[2] = {lds_addr_of(&As[0][r0 * LDP + c0]),
                            lds_addr_of(&As[1][r0 * LDP + c0])};
  const uint32_t ldsB[2] = {lds_addr_of(&Bs[0][r0 * LDP + c0]),
                            lds_addr_of(&Bs[1][r0 * LDP + c0])};
  const _Float16* gA = A  + (size_t)(m0 + r0) * Kr + c0;
  const _Float16* gB = Bt + (size_t)(n0 + r0) * Kr + c0;

  v8f acc[2][4];
#pragma unroll
  for (int i = 0; i < 2; ++i)
#pragma unroll
    for (int j = 0; j < 4; ++j)
#pragma unroll
      for (int r = 0; r < 8; ++r) acc[i][j][r] = 0.0f;

  // Prologue: stream tile 0 (4 async DMA instructions per wave).
  async_copy_b128(ldsA[0],     gA);
  async_copy_b128(ldsA[0] + 16, gA + 8);
  async_copy_b128(ldsB[0],     gB);
  async_copy_b128(ldsB[0] + 16, gB + 8);

  const int nsteps = Kr / BK;
  for (int step = 0; step < nsteps; ++step) {
    const int cur = step & 1;
    if (step + 1 < nsteps) {
      const int nxt = cur ^ 1;
      const size_t ko = (size_t)(step + 1) * BK;
      async_copy_b128(ldsA[nxt],      gA + ko);
      async_copy_b128(ldsA[nxt] + 16, gA + ko + 8);
      async_copy_b128(ldsB[nxt],      gB + ko);
      async_copy_b128(ldsB[nxt] + 16, gB + ko + 8);
      // ASYNCcnt completes in order: <=4 outstanding means tile `cur` landed.
      asm volatile("s_wait_asynccnt 0x4" ::: "memory");
    } else {
      asm volatile("s_wait_asynccnt 0x0" ::: "memory");
    }
    __syncthreads();   // make every wave's DMA results visible block-wide

    const _Float16* Ac = &As[cur][0];
    const _Float16* Bc = &Bs[cur][0];
    const v16h a0 = load_frag(&Ac[(wr * 32 + 0)  * LDP], LDP);
    const v16h a1 = load_frag(&Ac[(wr * 32 + 16) * LDP], LDP);
#pragma unroll
    for (int j = 0; j < 4; ++j) {
      const v16h bf = load_frag(&Bc[(wc * 64 + j * 16) * LDP], LDP);
      acc[0][j] = wmma_f16(a0, bf, acc[0][j]);
      acc[1][j] = wmma_f16(a1, bf, acc[1][j]);
    }
    __syncthreads();   // all reads of `cur` done before it is re-filled
  }

  // Epilogue. C layout: VGPR r, lane l -> (M = r + 8*(l>>4), N = l&15).
  const int mgrp = (lane >> 4) * 8;
#pragma unroll
  for (int i = 0; i < 2; ++i) {
    const int mrow = m0 + wr * 32 + i * 16 + mgrp;
#pragma unroll
    for (int j = 0; j < 4; ++j) {
      const int gn = n0 + wc * 64 + j * 16 + (lane & 15);
      const float bv = bias[gn];
#pragma unroll
      for (int r = 0; r < 8; ++r) {
        const int gm = mrow + r;
        float val = acc[i][j][r] + bv;
        if (mode == EP_QKV) {
          if (gn < Ee) {                      // q: fold attention scale
            outH[(size_t)gm * Nr + gn] = (_Float16)(val * QSCALE);
          } else if (gn < 2 * Ee) {           // k
            outH[(size_t)gm * Nr + gn] = (_Float16)val;
          } else {                            // v -> transposed vT[b,h,d,s]
            const int cj  = gn - 2 * Ee;
            const int hh  = cj >> 7, dd = cj & 127;
            const int bbi = gm >> 11, ssi = gm & (Ss - 1);
            vT[(((size_t)(bbi * Hh + hh) * HDd) + dd) * Ss + ssi] =
                (_Float16)val;
          }
        } else if (mode == EP_RES) {          // f32 residual add
          const size_t idx = (size_t)gm * Nr + gn;
          outF[idx] = res[idx] + val;
        } else {                              // GELU (gelu_new), f16 out
          const float x3 = val * val * val;
          const float t  = tanhf(GELU_C * (val + 0.044715f * x3));
          outH[(size_t)gm * Nr + gn] = (_Float16)(0.5f * val * (1.0f + t));
        }
      }
    }
  }
}

// ---------------------------------------------------------------------------
// Flash attention: one wave per 16-row q tile, 4 waves/block.
// scores = (q*scale) k^T with causal mask, online softmax, acc += P v.
// P goes C-layout -> LDS -> A-layout between the two WMMAs (s_wait_dscnt).
// ---------------------------------------------------------------------------
__global__ __launch_bounds__(128) void attn_kernel(
    const _Float16* __restrict__ qkv, const _Float16* __restrict__ vT,
    _Float16* __restrict__ attnO) {
  __shared__ _Float16 Pld[4][16 * LDP];
  const int lane = threadIdx.x & 31;
  const int wave = threadIdx.x >> 5;
  const int bh = blockIdx.x;                 // b*H + h
  const int bb = bh / Hh, hh = bh % Hh;
  const int q0 = (blockIdx.y * 4 + wave) * 16;

  const _Float16* qbase = qkv + ((size_t)bb * Ss) * E3 + (size_t)hh * HDd;
  const _Float16* kbase = qbase + Ee;
  const _Float16* vbase = vT + ((size_t)(bb * Hh + hh) * HDd) * Ss;
  _Float16* obase = attnO + ((size_t)bb * Ss) * Ee + (size_t)hh * HDd;
  _Float16* P = &Pld[wave][0];
  const int mgrp = (lane >> 4) * 8;

  // q fragments: 16 x 128 (4 K-steps of 32), q already scaled by 1/sqrt(HD).
  v16h qf[4];
#pragma unroll
  for (int t = 0; t < 4; ++t)
    qf[t] = load_frag(qbase + (size_t)q0 * E3 + t * 32, E3);

  v8f acc[8];
#pragma unroll
  for (int c = 0; c < 8; ++c)
#pragma unroll
    for (int r = 0; r < 8; ++r) acc[c][r] = 0.0f;
  float mrun[8], lrun[8];
#pragma unroll
  for (int r = 0; r < 8; ++r) { mrun[r] = -1e30f; lrun[r] = 0.0f; }

  for (int kb = 0; kb <= q0 + 15; kb += 32) {   // causal key range, chunks of 32
    // speculative prefetch of the next key chunk
    __builtin_prefetch(kbase + (size_t)(kb + 32 + (lane & 15)) * E3, 0, 0);
    v8f s0, s1;
#pragma unroll
    for (int r = 0; r < 8; ++r) { s0[r] = 0.0f; s1[r] = 0.0f; }
#pragma unroll
    for (int t = 0; t < 4; ++t) {
      const v16h kf0 = load_frag(kbase + (size_t)(kb + 0)  * E3 + t * 32, E3);
      const v16h kf1 = load_frag(kbase + (size_t)(kb + 16) * E3 + t * 32, E3);
      s0 = wmma_f16(qf[t], kf0, s0);
      s1 = wmma_f16(qf[t], kf1, s1);
    }
    // Online softmax per row (row reductions live in 16-lane halves).
#pragma unroll
    for (int r = 0; r < 8; ++r) {
      const int qi = q0 + r + mgrp;
      const int kc = kb + (lane & 15);
      float v0 = (kc      <= qi) ? s0[r] : -1e30f;
      float v1 = (kc + 16 <= qi) ? s1[r] : -1e30f;
      float mx = fmaxf(v0, v1);
#pragma unroll
      for (int d = 1; d < 16; d <<= 1) mx = fmaxf(mx, __shfl_xor(mx, d, 32));
      const float mi  = fmaxf(mrun[r], mx);
      const float scl = __expf(mrun[r] - mi);
      mrun[r] = mi;
      const float p0 = __expf(v0 - mi);
      const float p1 = __expf(v1 - mi);
      float rsum = p0 + p1;
#pragma unroll
      for (int d = 1; d < 16; d <<= 1) rsum += __shfl_xor(rsum, d, 32);
      lrun[r] = lrun[r] * scl + rsum;
#pragma unroll
      for (int c = 0; c < 8; ++c) acc[c][r] *= scl;
      const int mloc = r + mgrp;
      P[mloc * LDP + (lane & 15)]      = (_Float16)p0;
      P[mloc * LDP + 16 + (lane & 15)] = (_Float16)p1;
    }
    // C-layout -> A-layout through per-wave LDS; DS ops are in-order per wave,
    // the explicit wait pins data + compiler ordering.
    asm volatile("s_wait_dscnt 0x0" ::: "memory");
    const v16h pf = load_frag(P, LDP);
#pragma unroll
    for (int c = 0; c < 8; ++c) {
      const v16h vf = load_frag(vbase + (size_t)(c * 16) * Ss + kb, Ss);
      acc[c] = wmma_f16(pf, vf, acc[c]);
    }
  }
  // Normalize and store f16 attention output [b,s,h,d].
#pragma unroll
  for (int r = 0; r < 8; ++r) {
    const float inv = 1.0f / lrun[r];
    const int gm = q0 + r + mgrp;
#pragma unroll
    for (int c = 0; c < 8; ++c)
      obase[(size_t)gm * Ee + c * 16 + (lane & 15)] =
          (_Float16)(acc[c][r] * inv);
  }
}

// ---------------------------------------------------------------------------
// Launch
// ---------------------------------------------------------------------------
extern "C" void kernel_launch(void* const* d_in, const int* in_sizes, int n_in,
                              void* d_out, int out_size, void* d_ws,
                              size_t ws_size, hipStream_t stream) {
  (void)in_sizes; (void)n_in; (void)out_size; (void)ws_size;
  const float* x    = (const float*)d_in[0];
  const float* ln1g = (const float*)d_in[2];
  const float* ln1b = (const float*)d_in[3];
  const float* Wqkv = (const float*)d_in[4];
  const float* bqkv = (const float*)d_in[5];
  const float* Wo   = (const float*)d_in[6];
  const float* bo   = (const float*)d_in[7];
  const float* ln2g = (const float*)d_in[8];
  const float* ln2b = (const float*)d_in[9];
  const float* Wfc  = (const float*)d_in[10];
  const float* bfc  = (const float*)d_in[11];
  const float* Wp   = (const float*)d_in[12];
  const float* bp   = (const float*)d_in[13];
  float* out = (float*)d_out;

  uintptr_t p = (uintptr_t)d_ws;
  auto alloc = [&](size_t bytes) -> void* {
    p = (p + 255) & ~(uintptr_t)255;
    void* r = (void*)p;
    p += bytes;
    return r;
  };
  // Transposed f16 weights: Bt layout [N][K].
  _Float16* WqkvT = (_Float16*)alloc((size_t)E3 * Ee * 2);
  _Float16* WoT   = (_Float16*)alloc((size_t)Ee * Ee * 2);
  _Float16* WfcT  = (_Float16*)alloc((size_t)Mm * Ee * 2);
  _Float16* WpT   = (_Float16*)alloc((size_t)Ee * Mm * 2);
  _Float16* h1     = (_Float16*)alloc((size_t)TOK * Ee * 2);
  _Float16* qkv16  = (_Float16*)alloc((size_t)TOK * E3 * 2);
  _Float16* vTbuf  = (_Float16*)alloc((size_t)Bb * Hh * HDd * Ss * 2);
  _Float16* attn16 = (_Float16*)alloc((size_t)TOK * Ee * 2);
  float*    x1     = (float*)   alloc((size_t)TOK * Ee * 4);
  _Float16* h2     = (_Float16*)alloc((size_t)TOK * Ee * 2);
  _Float16* ff16   = (_Float16*)alloc((size_t)TOK * Mm * 2);

  // Weight conversion f32 -> f16 with transpose ([K][N] -> [N][K]).
  cvt_transpose<<<dim3(E3 / 32, Ee / 32), 256, 0, stream>>>(Wqkv, WqkvT, Ee, E3);
  cvt_transpose<<<dim3(Ee / 32, Ee / 32), 256, 0, stream>>>(Wo,   WoT,   Ee, Ee);
  cvt_transpose<<<dim3(Mm / 32, Ee / 32), 256, 0, stream>>>(Wfc,  WfcT,  Ee, Mm);
  cvt_transpose<<<dim3(Ee / 32, Mm / 32), 256, 0, stream>>>(Wp,   WpT,   Mm, Ee);

  // LN1 -> h1 (f16)
  ln_kernel<<<TOK, 256, 0, stream>>>(x, ln1g, ln1b, h1);

  // QKV GEMM (+q scale, +vT scatter)
  gemm_wmma_kernel<<<dim3(E3 / BN, TOK / BM), 256, 0, stream>>>(
      h1, WqkvT, bqkv, E3, Ee, EP_QKV, nullptr, nullptr, qkv16, vTbuf);

  // Flash attention -> attn16 (f16, [b,s,h,d])
  attn_kernel<<<dim3(Bb * Hh, Ss / 64), 128, 0, stream>>>(qkv16, vTbuf, attn16);

  // Output projection + residual -> x1 (f32)
  gemm_wmma_kernel<<<dim3(Ee / BN, TOK / BM), 256, 0, stream>>>(
      attn16, WoT, bo, Ee, Ee, EP_RES, x, x1, nullptr, nullptr);

  // LN2 -> h2 (f16)
  ln_kernel<<<TOK, 256, 0, stream>>>(x1, ln2g, ln2b, h2);

  // FFN fc + GELU -> ff16 (f16)
  gemm_wmma_kernel<<<dim3(Mm / BN, TOK / BM), 256, 0, stream>>>(
      h2, WfcT, bfc, Mm, Ee, EP_GELU, nullptr, nullptr, ff16, nullptr);

  // FFN proj + residual -> out (f32)
  gemm_wmma_kernel<<<dim3(Ee / BN, TOK / BM), 256, 0, stream>>>(
      ff16, WpT, bp, Ee, Mm, EP_RES, x1, out, nullptr, nullptr);
}
